// SaccadicLayer_16458314678649
// MI455X (gfx1250) — matrix operand order repaired and checked by reading, not compiled, not run on hardware
//
#include <hip/hip_runtime.h>

// ---------------------------------------------------------------- types
typedef _Float16 hf16;
typedef __attribute__((ext_vector_type(16))) _Float16 v16h;
typedef __attribute__((ext_vector_type(8)))  _Float16 v8h;
typedef __attribute__((ext_vector_type(8)))  float    v8f;

// ---------------------------------------------------------------- constants
#define CB   8        // batch
#define CN   4096     // seq len
#define CBD  2048     // full dim
#define CD   512      // model dim
#define CMID 256
#define CBS  64       // block size
#define CNB  64       // num blocks = N/BS
#define CWS  64       // window
#define CNS  4        // saccades
#define CTK  16       // top-k
#define CNH  8        // heads
#define CHD  64       // head dim

// Use GLOBAL_LOAD_ASYNC_TO_LDS_B128 for the (already-f16) B tile.
#define USE_ASYNC_LDS 1

__device__ __forceinline__ float gelu_f(float x) {
  return 0.5f * x * (1.0f + erff(x * 0.70710678118654752440f));
}

// ---------------------------------------------------------------- weight prep:
// f32 (K x N) row-major  ->  f16 (N x K) row-major (contiguous in K)
__global__ void prep_wt(const float* __restrict__ w, hf16* __restrict__ wt,
                        int Kd, int Nd) {
  long i = (long)blockIdx.x * 256 + threadIdx.x;
  if (i >= (long)Kd * Nd) return;
  int n = (int)(i / Kd);
  int k = (int)(i % Kd);
  wt[i] = (hf16)w[(long)k * Nd + n];
}

// conv weight (O=256, C=256, K=64) -> wt[o][k*256+c]   (O x 16384)
__global__ void prep_conv_wt(const float* __restrict__ w, hf16* __restrict__ wt) {
  long i = (long)blockIdx.x * 256 + threadIdx.x;
  if (i >= 256L * 16384) return;
  int o = (int)(i >> 14);
  int rem = (int)(i & 16383);
  int k = rem >> 8;
  int c = rem & 255;
  wt[i] = (hf16)w[((long)o * 256 + c) * 64 + k];
}

// ---------------------------------------------------------------- WMMA GEMM
// C[M x Nd] (f32, ldc) = act( A[M x Kd] (f32, lda) * Bt^T + bias ) (+C if OUTMODE)
// Bt is f16 (Nd x Kd), row-major (each output column's K-vector contiguous).
// Requires: M % 32 == 0, Nd % 128 == 0, Kd % 64 == 0.
// Block: 256 threads = 8 waves (2x4); block tile 32x128; wave tile 16x32.
// Double-buffered LDS, K-stage = 64 -> 4 WMMAs per barrier.
// B tile streamed via GLOBAL_LOAD_ASYNC_TO_LDS_B128 (ASYNCcnt); A tile via
// VGPR path with f32->f16 conversion.
template <int ACT, int OUTMODE>
__global__ __launch_bounds__(256)
void gemm_wmma(const float* __restrict__ A, long lda,
               const hf16* __restrict__ Bt,
               const float* __restrict__ bias,
               float* __restrict__ C, long ldc,
               int M, int Nd, int Kd) {
  __shared__ _Float16 As[2][32][72];
  __shared__ _Float16 Bs[2][128][72];
  const int tid  = threadIdx.x;
  const int lane = tid & 31;
  const int wave = tid >> 5;
  const int wrow = wave >> 2;    // 0..1
  const int wcol = wave & 3;     // 0..3
  const int mblk = blockIdx.y * 32;
  const int nblk = blockIdx.x * 128;

  // loader assignment
  const int ar = tid >> 3;            // 0..31  (A row)
  const int ac = (tid & 7) * 8;       // 0..56  (A k offset, 8 floats)
  const int bc = tid >> 1;            // 0..127 (B col)
  const int bk = (tid & 1) * 32;      // 0|32   (B k offset, 32 halves = 64B)

  const float* aptr = A + (long)(mblk + ar) * lda + ac;
  const hf16*  bptr = Bt + (long)(nblk + bc) * Kd + bk;

  auto load_stage = [&](int buf, int k0) {
#if USE_ASYNC_LDS
    {  // async: 64 contiguous bytes global -> LDS, no VGPR round-trip
      unsigned ldsb = (unsigned)(unsigned long)
          (__attribute__((address_space(3))) void*)&Bs[buf][bc][bk];
      unsigned long gb = (unsigned long)(bptr + k0);
      asm volatile(
          "global_load_async_to_lds_b128 %0, %1, off\n\t"
          "global_load_async_to_lds_b128 %0, %1, off offset:16\n\t"
          "global_load_async_to_lds_b128 %0, %1, off offset:32\n\t"
          "global_load_async_to_lds_b128 %0, %1, off offset:48"
          :: "v"(ldsb), "v"(gb) : "memory");
    }
#else
    {
      const hf16* bp = bptr + k0;
      v8h* bd = (v8h*)&Bs[buf][bc][bk];
      bd[0] = *(const v8h*)(bp + 0);
      bd[1] = *(const v8h*)(bp + 8);
      bd[2] = *(const v8h*)(bp + 16);
      bd[3] = *(const v8h*)(bp + 24);
    }
#endif
    float4 v0 = *(const float4*)(aptr + k0);
    float4 v1 = *(const float4*)(aptr + k0 + 4);
    if (k0 + 64 < Kd) {
      __builtin_prefetch((const void*)(aptr + k0 + 64), 0, 1);
      __builtin_prefetch((const void*)(bptr + k0 + 64), 0, 1);
    }
    v8h av;
    av[0] = (_Float16)v0.x; av[1] = (_Float16)v0.y;
    av[2] = (_Float16)v0.z; av[3] = (_Float16)v0.w;
    av[4] = (_Float16)v1.x; av[5] = (_Float16)v1.y;
    av[6] = (_Float16)v1.z; av[7] = (_Float16)v1.w;
    *(v8h*)&As[buf][ar][ac] = av;
  };

  v8f acc0 = {}, acc1 = {};
  const int nstage = Kd >> 6;
  load_stage(0, 0);
  for (int s = 0; s < nstage; ++s) {
#if USE_ASYNC_LDS
    asm volatile("s_wait_asynccnt 0x0" ::: "memory");
#endif
    __syncthreads();
    if (s + 1 < nstage) load_stage((s + 1) & 1, (s + 1) << 6);
    const int buf = s & 1;
    const _Float16* arp = &As[buf][wrow * 16 + (lane & 15)][0];
    const int asel = (lane < 16) ? 0 : 8;
    v16h af0, af1;
#pragma unroll
    for (int j = 0; j < 8; ++j) {
      af0[j]     = arp[asel + j];
      af0[j + 8] = arp[16 + asel + j];
      af1[j]     = arp[32 + asel + j];
      af1[j + 8] = arp[48 + asel + j];
    }
    const int bb = (lane < 16) ? 0 : 8;
    const _Float16* brp0 = &Bs[buf][wcol * 32 + (lane & 15)][0];
    const _Float16* brp1 = brp0 + 16 * 72;
    v16h b00, b10, b01, b11;
#pragma unroll
    for (int j = 0; j < 16; ++j) {
      b00[j] = brp0[bb + (j & 7) + ((j >> 3) << 4)];
      b10[j] = brp1[bb + (j & 7) + ((j >> 3) << 4)];
      b01[j] = brp0[32 + bb + (j & 7) + ((j >> 3) << 4)];
      b11[j] = brp1[32 + bb + (j & 7) + ((j >> 3) << 4)];
    }
    acc0 = __builtin_amdgcn_wmma_f32_16x16x32_f16(false, af0, false, b00,
                                                  (short)0, acc0, false, false);
    acc1 = __builtin_amdgcn_wmma_f32_16x16x32_f16(false, af0, false, b10,
                                                  (short)0, acc1, false, false);
    acc0 = __builtin_amdgcn_wmma_f32_16x16x32_f16(false, af1, false, b01,
                                                  (short)0, acc0, false, false);
    acc1 = __builtin_amdgcn_wmma_f32_16x16x32_f16(false, af1, false, b11,
                                                  (short)0, acc1, false, false);
  }
  const int ccol0 = nblk + wcol * 32 + (lane & 15);
  const int ccol1 = ccol0 + 16;
  const int rbase = mblk + wrow * 16 + ((lane < 16) ? 0 : 8);
  const float bv0 = bias[ccol0];
  const float bv1 = bias[ccol1];
#pragma unroll
  for (int r = 0; r < 8; ++r) {
    long i0 = (long)(rbase + r) * ldc + ccol0;
    long i1 = (long)(rbase + r) * ldc + ccol1;
    float u0 = acc0[r] + bv0;
    float u1 = acc1[r] + bv1;
    if (ACT == 1) { u0 = gelu_f(u0); u1 = gelu_f(u1); }
    if (OUTMODE == 1) { u0 += C[i0]; u1 += C[i1]; }
    C[i0] = u0;
    C[i1] = u1;
  }
}

// ---------------------------------------------------------------- LayerNorm (width 512)
__global__ __launch_bounds__(256)
void ln_kernel(const float* __restrict__ x, const float* __restrict__ g,
               const float* __restrict__ b, float* __restrict__ y,
               const float* __restrict__ pos, int posmod) {
  __shared__ float r1[256], r2[256];
  long row = blockIdx.x;
  int tid = threadIdx.x;
  const float* xr = x + row * CD;
  float v0 = xr[tid], v1 = xr[tid + 256];
  r1[tid] = v0 + v1;
  r2[tid] = v0 * v0 + v1 * v1;
  __syncthreads();
  for (int s = 128; s > 0; s >>= 1) {
    if (tid < s) { r1[tid] += r1[tid + s]; r2[tid] += r2[tid + s]; }
    __syncthreads();
  }
  float mean = r1[0] * (1.f / CD);
  float var  = r2[0] * (1.f / CD) - mean * mean;
  float rstd = rsqrtf(var + 1e-5f);
  float p0 = 0.f, p1 = 0.f;
  if (pos) {
    const float* pr = pos + (long)(row % posmod) * CD;
    p0 = pr[tid]; p1 = pr[tid + 256];
  }
  float* yr = y + row * CD;
  yr[tid]       = (v0 - mean) * rstd * g[tid]       + b[tid]       + p0;
  yr[tid + 256] = (v1 - mean) * rstd * g[tid + 256] + b[tid + 256] + p1;
}

// ---------------------------------------------------------------- block std(ddof=1)/max over 64 rows
__global__ __launch_bounds__(256)
void block_stats(const float* __restrict__ xf, float* __restrict__ stdo,
                 float* __restrict__ maxo) {
  long blk = blockIdx.x;
  const float* base = xf + blk * (long)CBS * CBD;
  for (int f = threadIdx.x; f < CBD; f += 256) {
    float s = 0.f, sq = 0.f, mx = -3.4e38f;
    for (int r = 0; r < CBS; ++r) {
      float v = base[(long)r * CBD + f];
      s += v; sq += v * v; mx = fmaxf(mx, v);
    }
    float mean = s * (1.f / CBS);
    float var = (sq - s * mean) * (1.f / (CBS - 1));
    stdo[blk * CBD + f] = sqrtf(fmaxf(var, 0.f));
    maxo[blk * CBD + f] = mx;
  }
}

// ---------------------------------------------------------------- state = mean over 64 map rows
__global__ void mean_state(const float* __restrict__ pmap, float* __restrict__ state) {
  int b = blockIdx.x, d = threadIdx.x;
  float s = 0.f;
  for (int m = 0; m < CNB; ++m) s += pmap[((long)b * CNB + m) * CD + d];
  state[b * CD + d] = s * (1.f / CNB);
}

// ---------------------------------------------------------------- naive small linear (M tiny)
__global__ void small_linear(const float* __restrict__ A, int lda,
                             const float* __restrict__ W, int ldw,
                             const float* __restrict__ bias,
                             float* __restrict__ C, int ldc,
                             int M, int Nd, int Kd) {
  long i = (long)blockIdx.x * 256 + threadIdx.x;
  if (i >= (long)M * Nd) return;
  int m = (int)(i / Nd), n = (int)(i % Nd);
  float s = bias[n];
  for (int k = 0; k < Kd; ++k) s += A[(long)m * lda + k] * W[(long)k * ldw + n];
  C[(long)m * ldc + n] = s;
}

// ---------------------------------------------------------------- controller scores
__global__ __launch_bounds__(256)
void scores_kernel(const float* __restrict__ qs, const float* __restrict__ ks,
                   float* __restrict__ sc) {
  __shared__ float red[256];
  int b = blockIdx.x >> 6;
  int tid = threadIdx.x;
  const float* q = qs + (long)b * CD;
  const float* k = ks + (long)blockIdx.x * CD;
  red[tid] = q[tid] * k[tid] + q[tid + 256] * k[tid + 256];
  __syncthreads();
  for (int s = 128; s > 0; s >>= 1) {
    if (tid < s) red[tid] += red[tid + s];
    __syncthreads();
  }
  if (tid == 0) sc[blockIdx.x] = red[0] * 0.04419417382415922f;  // 1/sqrt(512)
}

// ---------------------------------------------------------------- top-16 + softmax(v/5) + starts
__global__ void topk_kernel(const float* __restrict__ scores, int* __restrict__ idx,
                            float* __restrict__ wts, int* __restrict__ starts) {
  int b = blockIdx.x;
  if (threadIdx.x != 0) return;
  float s[CNB];
  for (int i = 0; i < CNB; ++i) s[i] = scores[b * CNB + i];
  int id[CTK]; float val[CTK];
  for (int k = 0; k < CTK; ++k) {
    float best = -3.4e38f; int bi = 0;
    for (int i = 0; i < CNB; ++i) if (s[i] > best) { best = s[i]; bi = i; }
    id[k] = bi; val[k] = best; s[bi] = -3.4e38f;
  }
  float mx = val[0], den = 0.f;
  float e[CTK];
  for (int k = 0; k < CTK; ++k) { e[k] = __expf((val[k] - mx) * 0.2f); den += e[k]; }
  for (int k = 0; k < CTK; ++k) {
    wts[b * CTK + k] = e[k] / den;
    idx[b * CTK + k] = id[k];
    int st = id[k] * CBS - CWS / 2;
    st = st < 0 ? 0 : (st > CN - CWS ? CN - CWS : st);
    starts[b * CTK + k] = st;
  }
}

// ---------------------------------------------------------------- build ctx = [state | acc | win]
__global__ void build_ctx(const float* __restrict__ state, const float* __restrict__ acc,
                          const float* __restrict__ h, const int* __restrict__ starts,
                          float* __restrict__ ctx, int L, int t) {
  int l = blockIdx.x, k = blockIdx.y, b = blockIdx.z, d = threadIdx.x;
  long dst = (((long)(b * CTK + k) * L + l) * CD) + d;
  float v;
  int nacc = t * CWS;
  if (l == 0) v = state[(long)b * CD + d];
  else if (l <= nacc) v = acc[((long)b * (CNS * CWS) + (l - 1)) * CD + d];
  else {
    int r = l - 1 - nacc;
    int s0 = starts[b * CTK + k];
    v = h[((long)b * CN + s0 + r) * CD + d];
  }
  ctx[dst] = v;
}

// ---------------------------------------------------------------- flash attention (online softmax)
__global__ __launch_bounds__(288)
void attn_kernel(const float* __restrict__ Q, long ldq, int qc0,
                 const float* __restrict__ Kb, long ldk, int kc0,
                 const float* __restrict__ Vb, long ldv, int vc0,
                 float* __restrict__ O, long ldo, int nq, int nkv) {
  __shared__ float kt[64][65];
  __shared__ float vt[64][65];
  int head = blockIdx.x & 7;
  int seq  = blockIdx.x >> 3;
  int tid  = threadIdx.x;
  bool active = tid < nq;
  float qreg[CHD], oacc[CHD];
  float mmax = -1e30f, ssum = 0.f;
  if (active) {
    const float* qrow = Q + ((long)seq * nq + tid) * ldq + qc0 + head * CHD;
#pragma unroll
    for (int d = 0; d < CHD; ++d) { qreg[d] = qrow[d]; oacc[d] = 0.f; }
  }
  for (int c0 = 0; c0 < nkv; c0 += 64) {
    int cn = nkv - c0; if (cn > 64) cn = 64;
    __syncthreads();
    for (int i = tid; i < cn * CHD; i += 288) {
      int j = i >> 6, d = i & 63;
      long r = (long)seq * nkv + c0 + j;
      kt[j][d] = Kb[r * ldk + kc0 + head * CHD + d];
      vt[j][d] = Vb[r * ldv + vc0 + head * CHD + d];
    }
    __syncthreads();
    if (active) {
      for (int j = 0; j < cn; ++j) {
        float s = 0.f;
#pragma unroll
        for (int d = 0; d < CHD; ++d) s += qreg[d] * kt[j][d];
        s *= 0.125f;
        float nm = fmaxf(mmax, s);
        float fs = __expf(mmax - nm);
        float e  = __expf(s - nm);
#pragma unroll
        for (int d = 0; d < CHD; ++d) oacc[d] = oacc[d] * fs + e * vt[j][d];
        ssum = ssum * fs + e;
        mmax = nm;
      }
    }
  }
  if (active) {
    float inv = 1.f / ssum;
    float* orow = O + ((long)seq * nq + tid) * ldo + head * CHD;
#pragma unroll
    for (int d = 0; d < CHD; ++d) orow[d] = oacc[d] * inv;
  }
}

// ---------------------------------------------------------------- misc small kernels
__global__ void gather_cls(const float* __restrict__ ctx, float* __restrict__ cls, int L) {
  int s = blockIdx.x, d = threadIdx.x;
  cls[(long)s * CD + d] = ctx[((long)s * L) * CD + d];
}

__global__ void state_update(const float* __restrict__ wts, const float* __restrict__ cls,
                             float* __restrict__ state) {
  int b = blockIdx.x, d = threadIdx.x;
  float s = 0.f;
  for (int k = 0; k < CTK; ++k)
    s += wts[b * CTK + k] * cls[((long)b * CTK + k) * CD + d];
  state[(long)b * CD + d] = s;
}

__global__ void acc_append(const float* __restrict__ h, const int* __restrict__ starts,
                           float* __restrict__ acc, int t) {
  int r = blockIdx.x, b = blockIdx.y, d = threadIdx.x;
  int s0 = starts[b * CTK + 0];
  acc[((long)b * (CNS * CWS) + t * CWS + r) * CD + d] =
      h[((long)b * CN + s0 + r) * CD + d];
}

__global__ __launch_bounds__(512)
void alpha_kernel(const float* __restrict__ g1w, const float* __restrict__ g1b,
                  const float* __restrict__ g2w, const float* __restrict__ g2b,
                  float* __restrict__ alpha, float tt) {
  __shared__ float red[512];
  int d = threadIdx.x;
  float u = tt * g1w[d] + g1b[d];
  red[d] = gelu_f(u) * g2w[d];
  __syncthreads();
  for (int s = 256; s > 0; s >>= 1) {
    if (d < s) red[d] += red[d + s];
    __syncthreads();
  }
  if (d == 0) {
    float v = red[0] + g2b[0];
    alpha[0] = 1.f / (1.f + __expf(-v));
  }
}

__global__ void axpy_kernel(float* __restrict__ y, const float* __restrict__ x,
                            const float* __restrict__ alpha, long n) {
  long i = (long)blockIdx.x * 256 + threadIdx.x;
  if (i < n) y[i] += alpha[0] * x[i];
}

__global__ void residual_broadcast(const float* __restrict__ xs, const float* __restrict__ so,
                                   float* __restrict__ xr) {
  long row = blockIdx.x;
  int d = threadIdx.x;
  int b = (int)(row >> 12);
  xr[row * CD + d] = xs[row * CD + d] + so[(long)b * CD + d];
}

// ---------------------------------------------------------------- host orchestration
extern "C" void kernel_launch(void* const* d_in, const int* in_sizes, int n_in,
                              void* d_out, int out_size, void* d_ws, size_t ws_size,
                              hipStream_t stream) {
  (void)in_sizes; (void)n_in; (void)out_size; (void)ws_size;
  const float* x_sacc = (const float*)d_in[0];
  const float* x_full = (const float*)d_in[1];

  char* ws = (char*)d_ws;
  size_t off = 0;
  auto allocf = [&](size_t n) {
    float* p = (float*)(ws + off);
    off = (off + n * sizeof(float) + 255) & ~(size_t)255;
    return p;
  };
  auto alloch = [&](size_t n) {
    hf16* p = (hf16*)(ws + off);
    off = (off + n * sizeof(hf16) + 255) & ~(size_t)255;
    return p;
  };
  auto alloci = [&](size_t n) {
    int* p = (int*)(ws + off);
    off = (off + n * sizeof(int) + 255) & ~(size_t)255;
    return p;
  };

  const long MN = (long)CB * CN;
  const int  Lmax = 1 + (CNS - 1) * CWS + CWS;  // 257

  float* h      = allocf(MN * CD);
  float* xres   = allocf(MN * CD);
  float* xmid   = allocf(MN * CMID);
  float* stdblk = allocf((long)CB * CNB * CBD);
  float* maxblk = allocf((long)CB * CNB * CBD);
  float* comb   = allocf((long)CB * CNB * 3 * CMID);
  float* praw   = allocf((long)CB * CNB * CD);
  float* pmap   = allocf((long)CB * CNB * CD);
  float* pmapln = allocf((long)CB * CNB * CD);
  float* state  = allocf(CB * CD);
  float* qs     = allocf(CB * CD);
  float* ks     = allocf((long)CB * CNB * CD);
  float* scores = allocf(CB * CNB);
  float* twts   = allocf(CB * CTK);
  int*   tidx   = alloci(CB * CTK);
  int*   tstart = alloci(CB * CTK);
  float* cls    = allocf(CB * CTK * CD);
  float* clsln  = allocf(CB * CTK * CD);
  float* ffnmid = allocf(CB * CTK * 4 * CD);
  float* accctx = allocf((long)CB * CNS * CWS * CD);
  float* qmap   = allocf((long)CB * CNB * CD);
  float* kvmap  = allocf((long)CB * CNS * CWS * 2 * CD);
  float* attnmp = allocf((long)CB * CNB * CD);
  float* delta  = allocf((long)CB * CNB * CD);
  float* alpha  = allocf(64);
  float* sout   = allocf(CB * CD);
  float* ctx    = allocf((long)CB * CTK * Lmax * CD);
  float* cn     = allocf((long)CB * CTK * Lmax * CD);
  float* qkv    = allocf((long)CB * CTK * Lmax * 3 * CD);
  float* attno  = allocf((long)CB * CTK * Lmax * CD);
  float* mid    = ctx;  // 32768 x 2048 aliases ctx+cn+qkv after the loop

  hf16* wt_convproj = alloch((long)CMID * CBD);
  hf16* wt_conv     = alloch(256L * 16384);
  hf16* wt_std      = alloch((long)CMID * CBD);
  hf16* wt_max      = alloch((long)CMID * CBD);
  hf16* wt_proj     = alloch((long)CD * 3 * CMID);
  hf16* wt_ctlk     = alloch((long)CD * CD);
  hf16* wt_fovin    = alloch(3L * CD * CD);
  hf16* wt_fovout   = alloch((long)CD * CD);
  hf16* wt_ffn1     = alloch(4L * CD * CD);
  hf16* wt_ffn2     = alloch(4L * CD * CD);
  hf16* wt_mapin    = alloch(3L * CD * CD);
  hf16* wt_mapout   = alloch((long)CD * CD);
  hf16* wt_mlp1     = alloch(4L * CD * CD);
  hf16* wt_mlp2     = alloch(4L * CD * CD);

  auto prep = [&](int wi, hf16* dst, int Kd, int Nd) {
    long n = (long)Kd * Nd;
    prep_wt<<<(unsigned)((n + 255) / 256), 256, 0, stream>>>((const float*)d_in[wi], dst, Kd, Nd);
  };
  prep(2, wt_convproj, CBD, CMID);
  prep_conv_wt<<<(unsigned)((256L * 16384 + 255) / 256), 256, 0, stream>>>((const float*)d_in[4], wt_conv);
  prep(6, wt_std, CBD, CMID);
  prep(8, wt_max, CBD, CMID);
  prep(10, wt_proj, 3 * CMID, CD);
  prep(17, wt_ctlk, CD, CD);
  prep(19, wt_fovin, CD, 3 * CD);
  prep(21, wt_fovout, CD, CD);
  prep(27, wt_ffn1, CD, 4 * CD);
  prep(29, wt_ffn2, 4 * CD, CD);
  prep(31, wt_mapin, CD, 3 * CD);
  prep(33, wt_mapout, CD, CD);
  prep(49, wt_mlp1, CD, 4 * CD);
  prep(51, wt_mlp2, 4 * CD, CD);

  // h = LN(x_sacc; ln1)
  ln_kernel<<<(unsigned)MN, 256, 0, stream>>>(x_sacc, (const float*)d_in[45], (const float*)d_in[46], h, nullptr, 1);

  // x_mid = x_full @ convproj + b   (32768 x 2048 -> 256)
  gemm_wmma<0, 0><<<dim3(CMID / 128, (unsigned)(MN / 32)), 256, 0, stream>>>(
      x_full, CBD, wt_convproj, (const float*)d_in[3], xmid, CMID, (int)MN, CMID, CBD);

  // conv_out -> comb[:, 0:256]   (512 x 16384 -> 256)
  gemm_wmma<0, 0><<<dim3(CMID / 128, (CB * CNB) / 32), 256, 0, stream>>>(
      xmid, (long)CBS * CMID, wt_conv, (const float*)d_in[5], comb, 3 * CMID,
      CB * CNB, CMID, CBS * CMID);

  // block std/max + projections
  block_stats<<<CB * CNB, 256, 0, stream>>>(x_full, stdblk, maxblk);
  gemm_wmma<0, 0><<<dim3(CMID / 128, (CB * CNB) / 32), 256, 0, stream>>>(
      stdblk, CBD, wt_std, (const float*)d_in[7], comb + CMID, 3 * CMID, CB * CNB, CMID, CBD);
  gemm_wmma<0, 0><<<dim3(CMID / 128, (CB * CNB) / 32), 256, 0, stream>>>(
      maxblk, CBD, wt_max, (const float*)d_in[9], comb + 2 * CMID, 3 * CMID, CB * CNB, CMID, CBD);

  // pmap = LN(comb @ project + b) + pos[:64]
  gemm_wmma<0, 0><<<dim3(CD / 128, (CB * CNB) / 32), 256, 0, stream>>>(
      comb, 3 * CMID, wt_proj, (const float*)d_in[11], praw, CD, CB * CNB, CD, 3 * CMID);
  ln_kernel<<<CB * CNB, 256, 0, stream>>>(praw, (const float*)d_in[12], (const float*)d_in[13],
                                          pmap, (const float*)d_in[14], CNB);
  mean_state<<<CB, CD, 0, stream>>>(pmap, state);

  for (int t = 0; t < CNS; ++t) {
    const int L = 1 + t * CWS + CWS;
    const int Mctx = CB * CTK * L;
    const int At = (t + 1) * CWS;

    // controller
    small_linear<<<(CB * CD + 255) / 256, 256, 0, stream>>>(
        state, CD, (const float*)d_in[15], CD, (const float*)d_in[16], qs, CD, CB, CD, CD);
    gemm_wmma<0, 0><<<dim3(CD / 128, (CB * CNB) / 32), 256, 0, stream>>>(
        pmap, CD, wt_ctlk, (const float*)d_in[18], ks, CD, CB * CNB, CD, CD);
    scores_kernel<<<CB * CNB, 256, 0, stream>>>(qs, ks, scores);
    topk_kernel<<<CB, 32, 0, stream>>>(scores, tidx, twts, tstart);

    // foveal context + self-attention block
    build_ctx<<<dim3(L, CTK, CB), CD, 0, stream>>>(state, accctx, h, tstart, ctx, L, t);
    ln_kernel<<<(unsigned)Mctx, 256, 0, stream>>>(ctx, (const float*)d_in[23], (const float*)d_in[24], cn, nullptr, 1);
    gemm_wmma<0, 0><<<dim3(3 * CD / 128, Mctx / 32), 256, 0, stream>>>(
        cn, CD, wt_fovin, (const float*)d_in[20], qkv, 3 * CD, Mctx, 3 * CD, CD);
    attn_kernel<<<CB * CTK * CNH, 288, 0, stream>>>(qkv, 3 * CD, 0, qkv, 3 * CD, CD,
                                                    qkv, 3 * CD, 2 * CD, attno, CD, L, L);
    gemm_wmma<0, 1><<<dim3(CD / 128, Mctx / 32), 256, 0, stream>>>(
        attno, CD, wt_fovout, (const float*)d_in[22], ctx, CD, Mctx, CD, CD);

    // cls FFN
    gather_cls<<<CB * CTK, CD, 0, stream>>>(ctx, cls, L);
    ln_kernel<<<CB * CTK, 256, 0, stream>>>(cls, (const float*)d_in[25], (const float*)d_in[26], clsln, nullptr, 1);
    gemm_wmma<1, 0><<<dim3(4 * CD / 128, (CB * CTK) / 32), 256, 0, stream>>>(
        clsln, CD, wt_ffn1, (const float*)d_in[28], ffnmid, 4 * CD, CB * CTK, 4 * CD, CD);
    gemm_wmma<0, 1><<<dim3(CD / 128, (CB * CTK) / 32), 256, 0, stream>>>(
        ffnmid, 4 * CD, wt_ffn2, (const float*)d_in[30], cls, CD, CB * CTK, CD, 4 * CD);

    state_update<<<CB, CD, 0, stream>>>(twts, cls, state);
    acc_append<<<dim3(CWS, CB), CD, 0, stream>>>(h, tstart, accctx, t);

    // map cross-attention
    ln_kernel<<<CB * CNB, 256, 0, stream>>>(pmap, (const float*)d_in[35], (const float*)d_in[36], pmapln, nullptr, 1);
    gemm_wmma<0, 0><<<dim3(CD / 128, (CB * CNB) / 32), 256, 0, stream>>>(
        pmapln, CD, wt_mapin, (const float*)d_in[32], qmap, CD, CB * CNB, CD, CD);
    for (int b = 0; b < CB; ++b) {
      gemm_wmma<0, 0><<<dim3(2 * CD / 128, At / 32), 256, 0, stream>>>(
          accctx + (long)b * CNS * CWS * CD, CD, wt_mapin + (long)CD * CD,
          (const float*)d_in[32] + CD, kvmap + (long)b * At * 2 * CD, 2 * CD,
          At, 2 * CD, CD);
    }
    attn_kernel<<<CB * CNH, 288, 0, stream>>>(qmap, CD, 0, kvmap, 2 * CD, 0,
                                              kvmap, 2 * CD, CD, attnmp, CD, CNB, At);
    gemm_wmma<0, 0><<<dim3(CD / 128, (CB * CNB) / 32), 256, 0, stream>>>(
        attnmp, CD, wt_mapout, (const float*)d_in[34], delta, CD, CB * CNB, CD, CD);

    alpha_kernel<<<1, CD, 0, stream>>>((const float*)d_in[37], (const float*)d_in[38],
                                       (const float*)d_in[39], (const float*)d_in[40],
                                       alpha, (float)t / (float)CNS);
    axpy_kernel<<<(unsigned)(((long)CB * CNB * CD + 255) / 256), 256, 0, stream>>>(
        pmap, delta, alpha, (long)CB * CNB * CD);
  }

  // out = LN(state; outn) @ outp + b, broadcast over N; x = residual + out
  ln_kernel<<<CB, 256, 0, stream>>>(state, (const float*)d_in[43], (const float*)d_in[44], qs, nullptr, 1);
  small_linear<<<(CB * CD + 255) / 256, 256, 0, stream>>>(
      qs, CD, (const float*)d_in[41], CD, (const float*)d_in[42], sout, CD, CB, CD, CD);
  residual_broadcast<<<(unsigned)MN, CD, 0, stream>>>(x_sacc, sout, xres);

  // final MLP: out = xres + GELU(LN(xres) @ mlp1 + b) @ mlp2 + b
  ln_kernel<<<(unsigned)MN, 256, 0, stream>>>(xres, (const float*)d_in[47], (const float*)d_in[48], h, nullptr, 1);
  gemm_wmma<1, 0><<<dim3(4 * CD / 128, (unsigned)(MN / 32)), 256, 0, stream>>>(
      h, CD, wt_mlp1, (const float*)d_in[50], mid, 4 * CD, (int)MN, 4 * CD, CD);
  hipMemcpyAsync(d_out, xres, (size_t)MN * CD * sizeof(float), hipMemcpyDeviceToDevice, stream);
  gemm_wmma<0, 1><<<dim3(CD / 128, (unsigned)(MN / 32)), 256, 0, stream>>>(
      mid, 4 * CD, wt_mlp2, (const float*)d_in[52], (float*)d_out, CD, (int)MN, CD, 4 * CD);
}
